// MoNet_10179072491909
// MI455X (gfx1250) — compile-verified, matrix operand order checked by reading, not run on hardware
//
#include <hip/hip_runtime.h>
#include <hip/hip_bf16.h>
#include <math.h>

// ---------- types for CDNA5 WMMA ----------
typedef __bf16 bf16_t;
typedef __attribute__((ext_vector_type(16))) __bf16 v16bf;
typedef __attribute__((ext_vector_type(8)))  __bf16 v8bf;
typedef __attribute__((ext_vector_type(8)))  float  v8f;

// ---------- LIF moment-activation constants ----------
#define LIF_L      0.05f
#define LIF_SQRTL  0.22360679774997896f
#define LIF_TREF   5.0f
#define LIF_VTHL   1.0f          /* VTH * L */
#define EPS_BN     1e-5f
#define EPS_SMALL  1e-10f

#define NB 8192                   /* batch (validated from in_sizes) */
#define NTAB 4001
#define TAB_SCALE (4000.0f/12.0f)

// =====================================================================
// Table construction:  g(x)=0.5*sqrt(pi)*e^{x^2}*erfc(-x) on [-6,6],
// G = cumtrapz(g), h = e^{x^2} * cumtrapz(e^{-x^2} g^2), H = cumtrapz(h)
// =====================================================================
__device__ void cumtrapz_block(const float* v, float* out, int N, float hdx, float* sh) {
    int t = threadIdx.x;
    const int T = 256;
    const int per = (N + T - 1) / T;      // 16
    int lo = t * per;
    int hi = lo + per; if (hi > N) hi = N;
    float sum = 0.0f;
    for (int k = (lo > 1 ? lo : 1); k < hi; ++k) sum += hdx * (v[k] + v[k - 1]);
    sh[t] = sum;
    __syncthreads();
    for (int off = 1; off < T; off <<= 1) {
        float a = (t >= off) ? sh[t - off] : 0.0f;
        __syncthreads();
        sh[t] += a;
        __syncthreads();
    }
    float run = (t == 0) ? 0.0f : sh[t - 1];
    for (int k = lo; k < hi; ++k) {
        if (k == 0) { out[0] = 0.0f; }
        else { run += hdx * (v[k] + v[k - 1]); out[k] = run; }
    }
    __syncthreads();
}

__global__ void build_tables_kernel(float* TG, float* TGI, float* THI,
                                    float* E, float* IT) {
    __shared__ float sh[256];
    const float dx = 12.0f / 4000.0f;
    int t = threadIdx.x;
    for (int k = t; k < NTAB; k += 256) {
        float x = -6.0f + k * dx;
        float g = 0.8862269254527580f * expf(x * x) * erfcf(-x);
        TG[k] = g;
        E[k]  = expf(-x * x) * g * g;
    }
    __syncthreads();
    cumtrapz_block(TG, TGI, NTAB, 0.5f * dx, sh);   // G
    cumtrapz_block(E,  IT,  NTAB, 0.5f * dx, sh);   // int e^{-x^2} g^2
    for (int k = t; k < NTAB; k += 256) {
        float x = -6.0f + k * dx;
        E[k] = expf(x * x) * IT[k];                 // h(x)
    }
    __syncthreads();
    cumtrapz_block(E, THI, NTAB, 0.5f * dx, sh);    // H
}

__device__ inline float tinterp(const float* tab, float x) {
    x = fminf(fmaxf(x, -6.0f), 6.0f);
    float p = (x + 6.0f) * TAB_SCALE;
    int j = (int)p;
    if (j > NTAB - 2) j = NTAB - 2;
    if (j < 0) j = 0;
    float f = p - (float)j;
    return tab[j] + f * (tab[j + 1] - tab[j]);
}

// =====================================================================
// WMMA fragment loader: rows are 64 bf16 wide, k contiguous.
//   A-layout (16-bit, 16x32): lane m=lane&15, half=lane>>4
//   elems 0..7  -> k = kbase + half*8 + 0..7
//   elems 8..15 -> k = kbase + 16 + half*8 + 0..7
//   (B operand uses the same pattern on row n; both ρ (symmetric) and
//    Aᵀ (read row-wise) make this a contiguous-row load.)
// =====================================================================
__device__ inline v16bf load_frag(const bf16_t* M, int row, int kbase, int half) {
    const v8bf* p0 = (const v8bf*)(M + row * 64 + kbase + half * 8);
    const v8bf* p1 = (const v8bf*)(M + row * 64 + kbase + 16 + half * 8);
    v8bf a = *p0, b = *p1;
    v16bf r;
#pragma unroll
    for (int j = 0; j < 8; ++j) { r[j] = a[j]; r[j + 8] = b[j]; }
    return r;
}

// D = Ahl * Bhl^ (3-term bf16 split, f32 accumulate), K = 64
__device__ inline v8f gemm_tile_bf16x3(const bf16_t* Ah, const bf16_t* Al,
                                       const bf16_t* Bh, const bf16_t* Bl,
                                       int mrow, int nrow, int half, v8f acc) {
#pragma unroll
    for (int kb = 0; kb < 64; kb += 32) {
        v16bf ah = load_frag(Ah, mrow, kb, half);
        v16bf al = load_frag(Al, mrow, kb, half);
        v16bf bh = load_frag(Bh, nrow, kb, half);
        v16bf bl = load_frag(Bl, nrow, kb, half);
        acc = __builtin_amdgcn_wmma_f32_16x16x32_bf16(false, ah, false, bh,
                                                      (short)0, acc, false, false);
        acc = __builtin_amdgcn_wmma_f32_16x16x32_bf16(false, ah, false, bl,
                                                      (short)0, acc, false, false);
        acc = __builtin_amdgcn_wmma_f32_16x16x32_bf16(false, al, false, bh,
                                                      (short)0, acc, false, false);
    }
    return acc;
}

// =====================================================================
// Middle layer (64 -> 64): one block per batch element.
//   A = W*diag(s);  T = A*rho;  C2 = T*Aᵀ   (bf16 hi/lo split WMMA)
//   also u2 = W*u + b and s2 = sqrt(clip(diag(C2))).
// =====================================================================
__global__ void __launch_bounds__(256)
layer_mid_wmma(const float* __restrict__ u_in, const float* __restrict__ s_in,
               const float* __restrict__ rho_in,
               const float* __restrict__ W, const float* __restrict__ bias,
               float* __restrict__ u2_buf, float* __restrict__ s2_buf,
               float* __restrict__ C2_out) {
    __shared__ bf16_t Ahi[4096], Alo[4096], Rhi[4096], Rlo[4096];
    __shared__ bf16_t Thi[4096], Tlo[4096];
    __shared__ float ss[64], uu[64];

    const int b = blockIdx.x;
    const int t = threadIdx.x;
    const int lane = t & 31;
    const int wave = t >> 5;
    const int half = lane >> 4;
    const int lx = lane & 15;
    const size_t base = (size_t)b * 4096;

    __builtin_prefetch(rho_in + base + t * 16, 0, 1);

    if (t < 64) { ss[t] = s_in[(size_t)b * 64 + t]; uu[t] = u_in[(size_t)b * 64 + t]; }
    __syncthreads();

    // Build A = W*diag(s) and R = rho in split-bf16 LDS (16 elems / thread)
    {
        const int row = t >> 2;
        const int cb = (t & 3) * 16;
#pragma unroll
        for (int j = 0; j < 16; ++j) {
            int c = cb + j;
            float a = W[row * 64 + c] * ss[c];
            bf16_t ah = (bf16_t)a;
            Ahi[row * 64 + c] = ah;
            Alo[row * 64 + c] = (bf16_t)(a - (float)ah);
            float r = rho_in[base + row * 64 + c];
            bf16_t rh = (bf16_t)r;
            Rhi[row * 64 + c] = rh;
            Rlo[row * 64 + c] = (bf16_t)(r - (float)rh);
        }
    }
    // u2 = W u + b (exact fp32)
    if (t < 64) {
        float acc = bias[t];
#pragma unroll 8
        for (int i = 0; i < 64; ++i) acc += W[t * 64 + i] * uu[i];
        u2_buf[(size_t)b * 64 + t] = acc;
    }
    __syncthreads();

    // GEMM1: T = A * rho    (B-frag = row n of symmetric rho)
#pragma unroll
    for (int ti = 0; ti < 2; ++ti) {
        const int tile = wave + ti * 8;
        const int tm = tile >> 2, tn = tile & 3;
        v8f acc = {};
        acc = gemm_tile_bf16x3(Ahi, Alo, Rhi, Rlo, tm * 16 + lx, tn * 16 + lx, half, acc);
#pragma unroll
        for (int r = 0; r < 8; ++r) {
            int m = tm * 16 + r + (half ? 8 : 0);
            int n = tn * 16 + lx;
            float v = acc[r];
            bf16_t h = (bf16_t)v;
            Thi[m * 64 + n] = h;
            Tlo[m * 64 + n] = (bf16_t)(v - (float)h);
        }
    }
    __syncthreads();

    // GEMM2: C2 = T * Aᵀ    (B-frag = row n of A)
#pragma unroll
    for (int ti = 0; ti < 2; ++ti) {
        const int tile = wave + ti * 8;
        const int tm = tile >> 2, tn = tile & 3;
        v8f acc = {};
        acc = gemm_tile_bf16x3(Thi, Tlo, Ahi, Alo, tm * 16 + lx, tn * 16 + lx, half, acc);
#pragma unroll
        for (int r = 0; r < 8; ++r) {
            int m = tm * 16 + r + (half ? 8 : 0);
            int n = tn * 16 + lx;
            C2_out[base + m * 64 + n] = acc[r];
        }
        if (tm == tn) {
#pragma unroll
            for (int r = 0; r < 8; ++r) {
                if (lane == r)
                    s2_buf[(size_t)b * 64 + tm * 16 + r] = sqrtf(fmaxf(acc[r], 1e-12f));
                if (lane == 24 + r)
                    s2_buf[(size_t)b * 64 + tm * 16 + 8 + r] = sqrtf(fmaxf(acc[r], 1e-12f));
            }
        }
    }
}

// =====================================================================
// First layer (2 -> 64): tiny K, plain VALU.
// =====================================================================
__global__ void layer_first(const float* __restrict__ u, const float* __restrict__ s,
                            const float* __restrict__ rho,
                            const float* __restrict__ W, const float* __restrict__ bias,
                            float* __restrict__ u2_buf, float* __restrict__ s2_buf,
                            float* __restrict__ C2) {
    __shared__ float a0[64], a1[64];
    __shared__ float cc[3], uv[2];
    const int b = blockIdx.x, t = threadIdx.x;
    if (t == 0) {
        float s0 = s[2 * b], s1 = s[2 * b + 1];
        float r01 = rho[4 * b + 1];
        cc[0] = s0 * s0; cc[1] = r01 * s0 * s1; cc[2] = s1 * s1;
        uv[0] = u[2 * b]; uv[1] = u[2 * b + 1];
    }
    if (t < 64) { a0[t] = W[t * 2]; a1[t] = W[t * 2 + 1]; }
    __syncthreads();
    if (t < 64)
        u2_buf[(size_t)b * 64 + t] = a0[t] * uv[0] + a1[t] * uv[1] + bias[t];
    const float c00 = cc[0], c01 = cc[1], c11 = cc[2];
    for (int idx = t; idx < 4096; idx += 256) {
        int o = idx >> 6, p = idx & 63;
        float v = a0[o] * a0[p] * c00 + (a0[o] * a1[p] + a1[o] * a0[p]) * c01
                + a1[o] * a1[p] * c11;
        C2[(size_t)b * 4096 + idx] = v;
        if (o == p) s2_buf[(size_t)b * 64 + o] = sqrtf(fmaxf(v, 1e-12f));
    }
}

// =====================================================================
// Last layer (64 -> 1): quadratic form + dot per batch element.
// =====================================================================
__global__ void layer_last(const float* __restrict__ u_in, const float* __restrict__ s_in,
                           const float* __restrict__ rho,
                           const float* __restrict__ W, const float* __restrict__ bias,
                           float* __restrict__ u2_buf, float* __restrict__ s2_buf) {
    __shared__ float a[64];
    __shared__ float red[256];
    const int b = blockIdx.x, t = threadIdx.x;
    if (t < 64) a[t] = W[t] * s_in[(size_t)b * 64 + t];
    __syncthreads();
    float part = 0.0f;
    for (int idx = t; idx < 4096; idx += 256) {
        int i = idx >> 6, j = idx & 63;
        part += a[i] * rho[(size_t)b * 4096 + idx] * a[j];
    }
    red[t] = part; __syncthreads();
    for (int off = 128; off; off >>= 1) { if (t < off) red[t] += red[t + off]; __syncthreads(); }
    if (t == 0) s2_buf[b] = sqrtf(fmaxf(red[0], 1e-12f));
    __syncthreads();
    float p2 = (t < 64) ? W[t] * u_in[(size_t)b * 64 + t] : 0.0f;
    red[t] = p2; __syncthreads();
    for (int off = 128; off; off >>= 1) { if (t < off) red[t] += red[t + off]; __syncthreads(); }
    if (t == 0) u2_buf[b] = red[0] + bias[0];
}

// =====================================================================
// BatchNorm statistics over the batch (two-pass, one block per feature)
// =====================================================================
__global__ void bn_stats(const float* __restrict__ u2, int B, int O,
                         float* __restrict__ mu, float* __restrict__ inv) {
    __shared__ float red[256];
    const int o = blockIdx.x, t = threadIdx.x;
    float s = 0.0f;
    for (int b = t; b < B; b += 256) s += u2[(size_t)b * O + o];
    red[t] = s; __syncthreads();
    for (int off = 128; off; off >>= 1) { if (t < off) red[t] += red[t + off]; __syncthreads(); }
    float m = red[0] / (float)B;
    __syncthreads();
    float v = 0.0f;
    for (int b = t; b < B; b += 256) { float d = u2[(size_t)b * O + o] - m; v += d * d; }
    red[t] = v; __syncthreads();
    for (int off = 128; off; off >>= 1) { if (t < off) red[t] += red[t + off]; __syncthreads(); }
    if (t == 0) { mu[o] = m; inv[o] = rsqrtf(red[0] / (float)B + EPS_BN); }
}

// =====================================================================
// Moment activation + in-place C2 -> rho rescale:
//   rho_out = set_diag_one( C2 ∘ f fᵀ ),  f = chi / s2
// =====================================================================
__global__ void __launch_bounds__(256)
act_kernel(const float* __restrict__ u2, const float* __restrict__ s2,
           const float* __restrict__ mu, const float* __restrict__ inv,
           const float* __restrict__ bn_w, const float* __restrict__ bn_b,
           const float* __restrict__ bns_w,
           const float* __restrict__ TG, const float* __restrict__ TGI,
           const float* __restrict__ THI,
           float* __restrict__ rho_io, float* __restrict__ u_out,
           float* __restrict__ s_out, int O) {
    __shared__ float f[64];
    const int b = blockIdx.x, t = threadIdx.x;
    if (t < O) {
        float x   = u2[(size_t)b * O + t];
        float ubn = (x - mu[t]) * inv[t] * bn_w[t] + bn_b[t];
        float s2v = s2[(size_t)b * O + t];
        float sbn = fmaxf(s2v * fabsf(bns_w[t] * inv[t]), EPS_SMALL);
        float isl = 1.0f / (sbn * LIF_SQRTL);
        float ub  = (LIF_VTHL - ubn) * isl;
        float lb  = (0.0f - ubn) * isl;
        float dG  = tinterp(TGI, ub) - tinterp(TGI, lb);
        float ua  = 1.0f / (LIF_TREF + (2.0f / LIF_L) * dG);
        float dH  = tinterp(THI, ub) - tinterp(THI, lb);
        float va  = (8.0f / (LIF_L * LIF_L)) * dH * ua * ua * ua;
        float sa  = sqrtf(fmaxf(va, 0.0f));
        float dg  = tinterp(TG, ub) - tinterp(TG, lb);
        float chi = (sa > EPS_SMALL)
                  ? ua * ua * (2.0f / (LIF_L * LIF_SQRTL)) * dg / fmaxf(sa, EPS_SMALL)
                  : 0.0f;
        u_out[(size_t)b * O + t] = ua;
        s_out[(size_t)b * O + t] = sa;
        f[t] = chi / s2v;
    }
    __syncthreads();
    const int OO = O * O;
    for (int idx = t; idx < OO; idx += 256) {
        int i = idx / O, j = idx - i * O;
        float v = (i == j) ? 1.0f : rho_io[(size_t)b * OO + idx] * f[i] * f[j];
        rho_io[(size_t)b * OO + idx] = v;
    }
}

// =====================================================================
// Host orchestration
// =====================================================================
extern "C" void kernel_launch(void* const* d_in, const int* in_sizes, int n_in,
                              void* d_out, int out_size, void* d_ws, size_t ws_size,
                              hipStream_t stream) {
    (void)n_in; (void)out_size; (void)ws_size;
    const int B = in_sizes[0] / 2;                 // u is [B, 2]
    const float* u0   = (const float*)d_in[0];
    const float* s0   = (const float*)d_in[1];
    const float* rho0 = (const float*)d_in[2];
    auto P = [&](int l, int k) { return (const float*)d_in[3 + 5 * l + k]; };
    const int NLAYER = 11;                         // 2 -> 64 x10 -> 1

    // ---- workspace layout (floats) ----
    float* wsf = (float*)d_ws;
    float* TG   = wsf + 0;
    float* TGI  = wsf + 4096;
    float* THI  = wsf + 8192;
    float* SCR1 = wsf + 12288;
    float* SCR2 = wsf + 16384;
    float* MU   = wsf + 20480;
    float* INV  = wsf + 20544;
    size_t off  = 20608;
    float* U2   = wsf + off; off += (size_t)B * 64;
    float* S2   = wsf + off; off += (size_t)B * 64;
    float* UA   = wsf + off; off += (size_t)B * 64;
    float* UB   = wsf + off; off += (size_t)B * 64;
    float* SA   = wsf + off; off += (size_t)B * 64;
    float* SB   = wsf + off; off += (size_t)B * 64;
    float* BIG0 = wsf + off; off += (size_t)B * 4096;
    float* BIG1 = wsf + off;

    build_tables_kernel<<<1, 256, 0, stream>>>(TG, TGI, THI, SCR1, SCR2);

    // ---- layer 0 : 2 -> 64 ----
    layer_first<<<B, 256, 0, stream>>>(u0, s0, rho0, P(0, 0), P(0, 1), U2, S2, BIG0);
    bn_stats<<<64, 256, 0, stream>>>(U2, B, 64, MU, INV);
    act_kernel<<<B, 256, 0, stream>>>(U2, S2, MU, INV, P(0, 2), P(0, 3), P(0, 4),
                                      TG, TGI, THI, BIG0, UA, SA, 64);

    float* ucur = UA; float* scur = SA; float* unx = UB; float* snx = SB;
    float* rcur = BIG0; float* rnx = BIG1;

    // ---- layers 1..9 : 64 -> 64 (WMMA bf16x3) ----
    for (int l = 1; l <= NLAYER - 2; ++l) {
        layer_mid_wmma<<<B, 256, 0, stream>>>(ucur, scur, rcur, P(l, 0), P(l, 1),
                                              U2, S2, rnx);
        bn_stats<<<64, 256, 0, stream>>>(U2, B, 64, MU, INV);
        act_kernel<<<B, 256, 0, stream>>>(U2, S2, MU, INV, P(l, 2), P(l, 3), P(l, 4),
                                          TG, TGI, THI, rnx, unx, snx, 64);
        float* tp;
        tp = ucur; ucur = unx; unx = tp;
        tp = scur; scur = snx; snx = tp;
        tp = rcur; rcur = rnx; rnx = tp;
    }

    // ---- layer 10 : 64 -> 1, outputs straight into d_out (u | s | rho) ----
    float* out = (float*)d_out;
    const int L = NLAYER - 1;
    layer_last<<<B, 256, 0, stream>>>(ucur, scur, rcur, P(L, 0), P(L, 1), U2, S2);
    bn_stats<<<1, 256, 0, stream>>>(U2, B, 1, MU, INV);
    act_kernel<<<B, 256, 0, stream>>>(U2, S2, MU, INV, P(L, 2), P(L, 3), P(L, 4),
                                      TG, TGI, THI,
                                      out + 2 * (size_t)B,   /* rho = all ones */
                                      out,                   /* u_a  */
                                      out + (size_t)B,       /* s_a  */
                                      1);
}
